// StrategicMoE_21955872817424
// MI455X (gfx1250) — matrix operand order repaired and checked by reading, not compiled, not run on hardware
//
#include <hip/hip_runtime.h>
#include <hip/hip_fp16.h>
#include <math.h>

typedef __attribute__((ext_vector_type(16))) _Float16 v16h;
typedef __attribute__((ext_vector_type(8)))  float    v8f;

#define B_TOT 65536
#define D_IN  44
#define H_DIM 256
#define N_EXP 4

// LDS layout (bytes)
#define SA_STRIDE 72    // halves, 144B rows (16B aligned, bank-spread)
#define SW_STRIDE 72
#define SH_STRIDE 264   // halves, 528B rows
#define OFF_SA  0
#define OFF_SW0 (128 * SA_STRIDE * 2)                  // 18432
#define OFF_SW1 (OFF_SW0 + 256 * SW_STRIDE * 2)        // 18432 + 36864 = 55296
#define OFF_SH  (OFF_SW1 + 256 * SW_STRIDE * 2)        // 92160
#define SMEM_BYTES (OFF_SH + 128 * SH_STRIDE * 2)      // + 67584 = 159744

// Workspace layout (bytes). Pre-swizzled f16 weight "LDS images".
#define PANEL256_H (256 * SW_STRIDE)   // 18432 halves = 36864 B
#define PANEL128_H (128 * SW_STRIDE)   // 9216 halves  = 18432 B
#define L3IMG_H    (16 * SH_STRIDE)    // 4224 halves  = 8448 B
#define PB256 (PANEL256_H * 2)
#define PB128 (PANEL128_H * 2)
#define PBL3  (L3IMG_H * 2)
#define WS_COMBINE 0
#define WS_EW1T (B_TOT * 4 * 4)                       // 1048576
#define WS_EW2T (WS_EW1T + 4 * PB256)
#define WS_EW3T (WS_EW2T + 16 * PB256)
#define WS_VW1T (WS_EW3T + 4 * PBL3)
#define WS_VW2T (WS_VW1T + PB256)
// total = WS_VW2T + 4*PB128 = 1930240 B (< 2 MB)

extern __shared__ char smem[];

typedef __attribute__((address_space(3))) char lds_char;

__device__ inline uint32_t lds_addr_of(const void* p) {
    return (uint32_t)(uintptr_t)(lds_char*)(void*)p;
}

// Issue an async global->LDS copy of BYTES contiguous bytes (ASYNCcnt path).
// Wave-chunked so every wave issues exactly ASYNC_N<BYTES> instructions
// (deterministic per-wave ASYNCcnt), 16B per lane per issue.
template <int BYTES>
struct AsyncN { static constexpr int value = ((BYTES / 8) + 511) / 512; };

template <int BYTES>
__device__ inline void issue_async_panel(void* sDst, const void* gSrc) {
    constexpr int CHUNK = BYTES / 8;             // bytes per wave (multiple of 16)
    constexpr int ITERS = AsyncN<BYTES>::value;
    const int wave = threadIdx.x >> 5;
    const int lane = threadIdx.x & 31;
    const uint32_t lds0 = lds_addr_of(sDst);
#pragma unroll
    for (int i = 0; i < ITERS; ++i) {
        if (lane * 16 + i * 512 < CHUNK) {
            int off = wave * CHUNK + lane * 16 + i * 512;
            uint32_t l = lds0 + (uint32_t)off;
            asm volatile("global_load_async_to_lds_b128 %0, %1, %2"
                         :: "v"(l), "v"(off), "s"(gSrc)
                         : "memory");
        }
    }
}

template <int N>
__device__ inline void wait_async() {
    asm volatile("s_wait_asynccnt %0" :: "i"(N) : "memory");
}

__device__ inline v8f zero8() {
    v8f z = {0.f, 0.f, 0.f, 0.f, 0.f, 0.f, 0.f, 0.f};
    return z;
}

// A fragment: 16x32 f16 (ISA 7.12.2): lane L: M=L%16, half=L/16;
// two contiguous 16B LDS loads at dword offsets 4h and 8+4h in the K window.
__device__ inline v16h load_frag_A(const _Float16* base, int strideH, int row16base, int kHalfBase) {
    const int lane = threadIdx.x & 31;
    const int m  = lane & 15;
    const int hh = lane >> 4;
    const uint4* p = (const uint4*)(base + (row16base + m) * strideH + kHalfBase);
    union { uint4 u[2]; v16h v; } U;
    U.u[0] = p[hh];
    U.u[1] = p[2 + hh];
    return U.v;
}

// B fragment: 32x16 f16, weights transposed in LDS as [N][strideH halves]:
// lane L: N=L%16; two contiguous 16B loads at dword offsets 4h and 8+4h.
__device__ inline v16h load_frag_B(const _Float16* wt, int strideH, int nBase, int kHalfBase) {
    const int lane = threadIdx.x & 31;
    const int n  = lane & 15;
    const int hh = lane >> 4;
    const uint4* p = (const uint4*)(wt + (nBase + n) * strideH + kHalfBase);
    union { uint4 u[2]; v16h v; } U;
    U.u[0] = p[hh];
    U.u[1] = p[2 + hh];
    return U.v;
}

// Double-buffered GEMM: [128 x K] (f16 LDS, stride As) x pre-swizzled panels.
// PRECONDITION: panel 0 already issued into sW0 by the caller (prefetch).
// Per-wave in-order async completion => s_wait_asynccnt NPER after issuing
// panel p+1 guarantees panel p is resident.
template <int KTILES, int NT, int NFULL>
__device__ void gemm_db(const _Float16* Asrc, int As, const _Float16* Wt,
                        _Float16* sW0, _Float16* sW1, _Float16* Hdst, int Hs) {
    constexpr int PBYTES = NFULL * SW_STRIDE * 2;
    constexpr int NPER = AsyncN<PBYTES>::value;
    const int tid  = threadIdx.x;
    const int wave = tid >> 5;
    const int lane = tid & 31;

    v8f acc[NT];
#pragma unroll
    for (int t = 0; t < NT; ++t) acc[t] = zero8();

    _Float16* bufs[2] = {sW0, sW1};
#pragma unroll
    for (int p = 0; p < KTILES; ++p) {
        if (p + 1 < KTILES)
            issue_async_panel<PBYTES>(bufs[(p + 1) & 1], Wt + (p + 1) * NFULL * SW_STRIDE);
        if (p + 1 < KTILES) wait_async<NPER>(); else wait_async<0>();
        __syncthreads();                       // panel p visible to all waves
        const _Float16* sW = bufs[p & 1];
#pragma unroll
        for (int ks = 0; ks < 2; ++ks) {
            v16h a = load_frag_A(Asrc, As, wave * 16, p * 64 + ks * 32);
#pragma unroll
            for (int t = 0; t < NT; ++t) {
                v16h b = load_frag_B(sW, SW_STRIDE, t * 16, ks * 32);
                acc[t] = __builtin_amdgcn_wmma_f32_16x16x32_f16(
                    false, a, false, b, (short)0, acc[t], false, false);
            }
        }
        __syncthreads();                       // all waves done with this buffer
    }
    const int n0 = lane & 15;
    const int hh = lane >> 4;
#pragma unroll
    for (int t = 0; t < NT; ++t)
#pragma unroll
        for (int r = 0; r < 8; ++r)
            Hdst[(wave * 16 + r + 8 * hh) * Hs + t * 16 + n0] = (_Float16)acc[t][r];
}

// Bias + optional LayerNorm + ReLU, in place. 2 threads per row, b128 LDS ops,
// partial sums combined with an intra-wave shuffle (wave32; lanes t,t^1 pair).
__device__ void ln_relu(_Float16* Hbuf, int Hs, int N,
                        const float* bias, const float* g, const float* bt, bool doLN) {
    __syncthreads();
    const int tid  = threadIdx.x;
    const int row  = tid >> 1;
    const int half = tid & 1;
    const int C = N >> 1;                      // cols per thread
    const int V = C >> 3;                      // uint4 chunks per thread
    _Float16* rp = Hbuf + row * Hs + half * C;
    union { uint4 u; _Float16 h[8]; } U;
    float m = 0.f, inv = 0.f;
    if (doLN) {
        float s = 0.f, s2 = 0.f;
        for (int v = 0; v < V; ++v) {
            U.u = ((const uint4*)rp)[v];
#pragma unroll
            for (int i = 0; i < 8; ++i) {
                float x = (float)U.h[i] + bias[half * C + v * 8 + i];
                s += x; s2 += x * x;
            }
        }
        s  += __shfl_xor(s, 1);
        s2 += __shfl_xor(s2, 1);
        m = s / N;
        float var = s2 / N - m * m;
        inv = rsqrtf(var + 1e-5f);
    }
    for (int v = 0; v < V; ++v) {
        U.u = ((const uint4*)rp)[v];
#pragma unroll
        for (int i = 0; i < 8; ++i) {
            int j = half * C + v * 8 + i;
            float x = (float)U.h[i] + bias[j];
            float y = doLN ? ((x - m) * inv * g[j] + bt[j]) : x;
            U.h[i] = (_Float16)fmaxf(y, 0.f);
        }
        ((uint4*)rp)[v] = U.u;
    }
    __syncthreads();
}

// One-time weight pre-swizzle: f32 row-major -> f16 transposed, K-padded
// "LDS image" panels (byte-exact for the async copies above).
__global__ __launch_bounds__(256) void prep_weights(
    const float* __restrict__ eW1, const float* __restrict__ eW2,
    const float* __restrict__ eW3, const float* __restrict__ vW1,
    const float* __restrict__ vW2, char* __restrict__ ws) {
    const int gid = blockIdx.x * 256 + threadIdx.x;
    const int gsz = gridDim.x * 256;
    _Float16* eW1t = (_Float16*)(ws + WS_EW1T);
    _Float16* eW2t = (_Float16*)(ws + WS_EW2T);
    _Float16* eW3t = (_Float16*)(ws + WS_EW3T);
    _Float16* vW1t = (_Float16*)(ws + WS_VW1T);
    _Float16* vW2t = (_Float16*)(ws + WS_VW2T);

    for (int i = gid; i < 4 * PANEL256_H; i += gsz) {
        int e = i / PANEL256_H, rr = i % PANEL256_H;
        int n = rr / SW_STRIDE, k = rr % SW_STRIDE;
        eW1t[i] = (_Float16)((k < D_IN) ? eW1[(e * D_IN + k) * H_DIM + n] : 0.f);
    }
    for (int i = gid; i < 16 * PANEL256_H; i += gsz) {
        int e = i / (4 * PANEL256_H), rem = i % (4 * PANEL256_H);
        int p = rem / PANEL256_H, rr = rem % PANEL256_H;
        int n = rr / SW_STRIDE, k = rr % SW_STRIDE;
        eW2t[i] = (_Float16)((k < 64) ? eW2[(e * H_DIM + p * 64 + k) * H_DIM + n] : 0.f);
    }
    for (int i = gid; i < 4 * L3IMG_H; i += gsz) {
        int e = i / L3IMG_H, rr = i % L3IMG_H;
        int n = rr / SH_STRIDE, k = rr % SH_STRIDE;
        eW3t[i] = (_Float16)((k < H_DIM) ? eW3[e * H_DIM * 16 + k * 16 + n] : 0.f);
    }
    for (int i = gid; i < PANEL256_H; i += gsz) {
        int n = i / SW_STRIDE, k = i % SW_STRIDE;
        vW1t[i] = (_Float16)((k < D_IN) ? vW1[k * H_DIM + n] : 0.f);
    }
    for (int i = gid; i < 4 * PANEL128_H; i += gsz) {
        int p = i / PANEL128_H, rr = i % PANEL128_H;
        int n = rr / SW_STRIDE, k = rr % SW_STRIDE;
        vW2t[i] = (_Float16)((k < 64) ? vW2[(p * 64 + k) * 128 + n] : 0.f);
    }
}

__global__ __launch_bounds__(256) void moe_kernel(
    const float* __restrict__ state,
    const _Float16* __restrict__ eW1t, const float* __restrict__ eb1,
    const float* __restrict__ eg1, const float* __restrict__ ebt1,
    const _Float16* __restrict__ eW2t, const float* __restrict__ eb2,
    const float* __restrict__ eg2, const float* __restrict__ ebt2,
    const _Float16* __restrict__ eW3t, const float* __restrict__ eb3,
    const _Float16* __restrict__ vW1t, const float* __restrict__ vb1,
    const float* __restrict__ vg,  const float* __restrict__ vbt,
    const _Float16* __restrict__ vW2t, const float* __restrict__ vb2,
    const float* __restrict__ vW3, const float* __restrict__ vb3,
    const float* __restrict__ combine, float* __restrict__ out) {
    const int tid  = threadIdx.x;
    const int wave = tid >> 5;
    const int lane = tid & 31;
    const int b0   = blockIdx.x * 128;

    _Float16* sA  = (_Float16*)(smem + OFF_SA);   // [128][72] state f16
    _Float16* sW0 = (_Float16*)(smem + OFF_SW0);  // panel buffer 0
    _Float16* sW1 = (_Float16*)(smem + OFF_SW1);  // panel buffer 1
    _Float16* sH  = (_Float16*)(smem + OFF_SH);   // [128][264] activations

    // prefetch expert 0 layer-1 weights while we stage the state tile
    issue_async_panel<PB256>(sW0, eW1t);

    for (int idx = tid; idx < 128 * 64; idx += 256) {
        int k = idx & 63;
        int r = idx >> 6;
        float x = (k < D_IN) ? state[(size_t)(b0 + r) * D_IN + k] : 0.f;
        sA[r * SA_STRIDE + k] = (_Float16)x;
    }

    v8f mix = zero8();

    for (int e = 0; e < N_EXP; ++e) {
        // layer1: [128,44] x [44,256] -> sH   (panel already in flight -> sW0)
        gemm_db<1, 16, 256>(sA, SA_STRIDE, eW1t + e * PANEL256_H, sW0, sW1, sH, SH_STRIDE);
        issue_async_panel<PB256>(sW0, eW2t + e * 4 * PANEL256_H);   // L2 panel 0
        ln_relu(sH, SH_STRIDE, H_DIM, eb1 + e * H_DIM, eg1 + e * H_DIM, ebt1 + e * H_DIM, true);
        // layer2: [128,256] x [256,256] -> sH (in place; barriers protect)
        gemm_db<4, 16, 256>(sH, SH_STRIDE, eW2t + e * 4 * PANEL256_H, sW0, sW1, sH, SH_STRIDE);
        issue_async_panel<PBL3>(sW0, eW3t + e * L3IMG_H);           // L3 image
        ln_relu(sH, SH_STRIDE, H_DIM, eb2 + e * H_DIM, eg2 + e * H_DIM, ebt2 + e * H_DIM, true);

        // layer3: [128,256] x [256,16]; image [16][264] in sW0
        wait_async<0>();
        __syncthreads();
        v8f o = zero8();
#pragma unroll
        for (int ks = 0; ks < 8; ++ks) {
            v16h a = load_frag_A(sH, SH_STRIDE, wave * 16, ks * 32);
            v16h b = load_frag_B(sW0, SH_STRIDE, 0, ks * 32);
            o = __builtin_amdgcn_wmma_f32_16x16x32_f16(false, a, false, b, (short)0, o,
                                                       false, false);
        }
        __syncthreads();
        // prefetch next layer-1 panel (or the value head's) into sW0
        if (e + 1 < N_EXP) issue_async_panel<PB256>(sW0, eW1t + (e + 1) * PANEL256_H);
        else               issue_async_panel<PB256>(sW0, vW1t);

        // fold into mixture with gate weight
        const int n0 = lane & 15;
        const int hh = lane >> 4;
#pragma unroll
        for (int r = 0; r < 8; ++r) {
            int bidx = b0 + wave * 16 + r + 8 * hh;
            float val = o[r] + eb3[e * 16 + n0];
            mix[r] += combine[(size_t)bidx * 4 + e] * val;
        }
    }

    // alpha / beta epilogue: softplus(x)+1
    {
        const int n0 = lane & 15;
        const int hh = lane >> 4;
#pragma unroll
        for (int r = 0; r < 8; ++r) {
            int bidx = b0 + wave * 16 + r + 8 * hh;
            float x = mix[r];
            float sp = (x > 20.f) ? x : log1pf(expf(x));
            float res = sp + 1.f;
            if (n0 < 8)
                out[(size_t)bidx * 8 + n0] = res;                            // alpha
            else
                out[(size_t)B_TOT * 8 + (size_t)bidx * 8 + (n0 - 8)] = res;  // beta
        }
    }

    // ---- value head ----
    gemm_db<1, 16, 256>(sA, SA_STRIDE, vW1t, sW0, sW1, sH, SH_STRIDE);
    issue_async_panel<PB128>(sW0, vW2t);
    ln_relu(sH, SH_STRIDE, H_DIM, vb1, vg, vbt, true);
    gemm_db<4, 8, 128>(sH, SH_STRIDE, vW2t, sW0, sW1, sH, SH_STRIDE);
    ln_relu(sH, SH_STRIDE, 128, vb2, nullptr, nullptr, false);
    if (tid < 128) {
        float acc = vb3[0];
        const uint4* rp = (const uint4*)(sH + tid * SH_STRIDE);
        for (int v = 0; v < 16; ++v) {
            union { uint4 u; _Float16 h[8]; } U;
            U.u = rp[v];
#pragma unroll
            for (int i = 0; i < 8; ++i) acc += (float)U.h[i] * vW3[v * 8 + i];
        }
        out[(size_t)B_TOT * 16 + (b0 + tid)] = acc;
    }
}

__global__ __launch_bounds__(256) void router_kernel(
    const float* __restrict__ state,
    const float* __restrict__ rW1, const float* __restrict__ rb1,
    const float* __restrict__ rW2, const float* __restrict__ rb2,
    float* __restrict__ out, float* __restrict__ combine) {
    __shared__ float sW1[D_IN * 128];
    __shared__ float sb1[128];
    __shared__ float sW2[128 * 4];
    __shared__ float sb2[4];
    const int tid = threadIdx.x;
    for (int i = tid; i < D_IN * 128; i += 256) sW1[i] = rW1[i];
    for (int i = tid; i < 128; i += 256) sb1[i] = rb1[i];
    for (int i = tid; i < 512; i += 256) sW2[i] = rW2[i];
    if (tid < 4) sb2[tid] = rb2[tid];
    __syncthreads();

    const int b = blockIdx.x * 256 + tid;
    float st[D_IN];
#pragma unroll
    for (int i = 0; i < D_IN; ++i) st[i] = state[(size_t)b * D_IN + i];

    float logit[4] = {sb2[0], sb2[1], sb2[2], sb2[3]};
    for (int j = 0; j < 128; ++j) {
        float h = sb1[j];
#pragma unroll
        for (int i = 0; i < D_IN; ++i) h += st[i] * sW1[i * 128 + j];
        h = fmaxf(h, 0.f);
#pragma unroll
        for (int c = 0; c < 4; ++c) logit[c] += h * sW2[j * 4 + c];
    }
    float mx = fmaxf(fmaxf(logit[0], logit[1]), fmaxf(logit[2], logit[3]));
    float p[4], s = 0.f;
#pragma unroll
    for (int c = 0; c < 4; ++c) { p[c] = expf(logit[c] - mx); s += p[c]; }
#pragma unroll
    for (int c = 0; c < 4; ++c) p[c] /= s;
#pragma unroll
    for (int c = 0; c < 4; ++c)
        out[(size_t)B_TOT * 17 + (size_t)b * 4 + c] = p[c];

    int i1 = 0;
    for (int c = 1; c < 4; ++c) if (p[c] > p[i1]) i1 = c;
    int i2 = -1;
    for (int c = 0; c < 4; ++c) {
        if (c == i1) continue;
        if (i2 < 0 || p[c] > p[i2]) i2 = c;
    }
    float vsum = p[i1] + p[i2] + 1e-8f;
    float cw[4] = {0.f, 0.f, 0.f, 0.f};
    cw[i1] = p[i1] / vsum;
    cw[i2] = p[i2] / vsum;
#pragma unroll
    for (int c = 0; c < 4; ++c) combine[(size_t)b * 4 + c] = cw[c];
}

extern "C" void kernel_launch(void* const* d_in, const int* in_sizes, int n_in,
                              void* d_out, int out_size, void* d_ws, size_t ws_size,
                              hipStream_t stream) {
    const float* state = (const float*)d_in[0];
    const float* rW1 = (const float*)d_in[1];
    const float* rb1 = (const float*)d_in[2];
    const float* rW2 = (const float*)d_in[3];
    const float* rb2 = (const float*)d_in[4];
    const float* eW1 = (const float*)d_in[5];
    const float* eb1 = (const float*)d_in[6];
    const float* eg1 = (const float*)d_in[7];
    const float* ebt1 = (const float*)d_in[8];
    const float* eW2 = (const float*)d_in[9];
    const float* eb2 = (const float*)d_in[10];
    const float* eg2 = (const float*)d_in[11];
    const float* ebt2 = (const float*)d_in[12];
    const float* eW3 = (const float*)d_in[13];
    const float* eb3 = (const float*)d_in[14];
    const float* vW1 = (const float*)d_in[15];
    const float* vb1 = (const float*)d_in[16];
    const float* vg = (const float*)d_in[17];
    const float* vbt = (const float*)d_in[18];
    const float* vW2 = (const float*)d_in[19];
    const float* vb2 = (const float*)d_in[20];
    const float* vW3 = (const float*)d_in[21];
    const float* vb3 = (const float*)d_in[22];
    float* out = (float*)d_out;

    char* ws = (char*)d_ws;
    float* combine = (float*)(ws + WS_COMBINE);  // [B,4] f32 = 1 MB

    prep_weights<<<512, 256, 0, stream>>>(eW1, eW2, eW3, vW1, vW2, ws);
    router_kernel<<<B_TOT / 256, 256, 0, stream>>>(state, rW1, rb1, rW2, rb2, out, combine);
    moe_kernel<<<B_TOT / 128, 256, SMEM_BYTES, stream>>>(
        state,
        (const _Float16*)(ws + WS_EW1T), eb1, eg1, ebt1,
        (const _Float16*)(ws + WS_EW2T), eb2, eg2, ebt2,
        (const _Float16*)(ws + WS_EW3T), eb3,
        (const _Float16*)(ws + WS_VW1T), vb1, vg, vbt,
        (const _Float16*)(ws + WS_VW2T), vb2,
        vW3, vb3, combine, out);
}
